// RNN_63213328662812
// MI455X (gfx1250) — matrix operand order repaired and checked by reading, not compiled, not run on hardware
//
#include <hip/hip_runtime.h>
#include <hip/hip_bf16.h>
#include <stdint.h>

#define B_  16
#define L_  512
#define D_  1024
#define NL_ 4
#define M_  (B_ * L_)     // 8192
#define SCAN_NWG 64

typedef __attribute__((ext_vector_type(16))) __bf16    v16bf;
typedef __attribute__((ext_vector_type(8)))  float     v8f;
typedef __attribute__((ext_vector_type(4)))  uint32_t  v4u;
typedef __attribute__((ext_vector_type(8)))  int       v8i;
typedef __attribute__((ext_vector_type(4)))  int       v4i;

union Frag16 {
  uint32_t u[8];
  v16bf    v;
};

__device__ __forceinline__ v8f wmma_bf16(v16bf a, v16bf b, v8f c) {
  // D = A(16x32 bf16) * B(32x16 bf16) + C(16x16 f32)
  return __builtin_amdgcn_wmma_f32_16x16x32_bf16(
      /*neg_a=*/false, a, /*neg_b=*/false, b,
      /*c_mod=*/(short)0, c, /*reuse_a=*/false, /*reuse_b=*/false);
}

// Low 32 bits of a generic LDS-aperture address are the LDS byte offset
// (flat aperture mapping).
__device__ __forceinline__ uint32_t lds_offset(const void* p) {
  return (uint32_t)(uintptr_t)p;
}

// ---------------------------------------------------------------------------
// Tensor Data Mover: 2D tile load Global -> LDS (ISA ch.8 D# layout).
//   bf16 elements (data_size=1 -> 2 bytes). tile_d0 = contiguous elements,
//   tile_d1 = rows, stride_d0 = row stride in elements. Tracked by TENSORcnt.
//   amdgpu-toolchain (clang-23) 6-arg builtin form.
// ---------------------------------------------------------------------------
__device__ __forceinline__ void tdm_load_2d_bf16(
    uint32_t lds_byte_off, const void* gaddr,
    uint32_t tile_d0, uint32_t tile_d1,
    uint32_t tensor_d0, uint32_t tensor_d1, uint32_t stride_d0) {
  const uint64_t ga = (uint64_t)(uintptr_t)gaddr;
  v4u g0;
  g0[0] = 1u;                                        // count=1, user-mode, no gather
  g0[1] = lds_byte_off;                              // lds_addr  [63:32]
  g0[2] = (uint32_t)ga;                              // global_addr[31:0]  -> bits [95:64]
  g0[3] = ((uint32_t)(ga >> 32) & 0x01FFFFFFu)       // global_addr[56:32] -> bits [120:96]
        | (2u << 30);                                // type=2 ("image")   -> bits [127:126]
  v8i g1;
  g1[0] = (int)(1u << 16);                           // workgroup_mask=0, data_size=1 (2B)
  g1[1] = (int)((tensor_d0 & 0xFFFFu) << 16);        // tensor_dim0[15:0] @ [63:48]
  g1[2] = (int)(((tensor_d0 >> 16) & 0xFFFFu)        // tensor_dim0[31:16] @ [79:64]
        |        ((tensor_d1 & 0xFFFFu) << 16));     // tensor_dim1[15:0]  @ [95:80]
  g1[3] = (int)(((tensor_d1 >> 16) & 0xFFFFu)        // tensor_dim1[31:16] @ [111:96]
        |        ((tile_d0 & 0xFFFFu) << 16));       // tile_dim0          @ [127:112]
  g1[4] = (int)(tile_d1 & 0xFFFFu);                  // tile_dim1; tile_dim2=0
  g1[5] = (int)stride_d0;                            // tensor_dim0_stride[31:0]
  g1[6] = 0;                                         // stride hi / dim1_stride
  g1[7] = 0;
  const v4i z4 = {0, 0, 0, 0};                       // 2D: groups 2/3 unused
  const v8i z8 = {0, 0, 0, 0, 0, 0, 0, 0};           // unused extra group (clang-23 form)
  __builtin_amdgcn_tensor_load_to_lds(g0, g1, z4, z4, z8, 0);
}

// A fragment: 16x32 bf16 tile, rows m0..m0+15, cols kk..kk+31, row stride lda.
// Lanes 0-15 hold M=lane, K=0..15; lanes 16-31 hold K=16..31 (matches B frags).
__device__ __forceinline__ v16bf load_a_frag(const __hip_bfloat16* A, int lda,
                                             int m0, int kk, int lane) {
  const int mr = lane & 15;
  const int kb = (lane >> 4) << 4;    // 0 or 16
  const __hip_bfloat16* r = A + (size_t)(m0 + mr) * lda + kk;
  Frag16 f;
  *reinterpret_cast<uint4*>(&f.u[0]) = *reinterpret_cast<const uint4*>(r + kb);
  *reinterpret_cast<uint4*>(&f.u[4]) = *reinterpret_cast<const uint4*>(r + kb + 8);
  return f.v;
}

// B fragment: 32x16 (KxN) with B[k][n] = Wt[(e0+n)*ldw + kk + k]
// (weights stored row-major (N,K)). Lanes 0-15: N=lane, K=0..15; lanes 16-31: K=16..31.
__device__ __forceinline__ v16bf load_b_frag(const __hip_bfloat16* Wt, int ldw,
                                             int e0, int kk, int lane) {
  const int n  = lane & 15;
  const int kh = (lane >> 4) << 4;    // 0 or 16
  const __hip_bfloat16* r = Wt + (size_t)(e0 + n) * ldw + kk + kh;
  Frag16 f;
  *reinterpret_cast<uint4*>(&f.u[0]) = *reinterpret_cast<const uint4*>(r);
  *reinterpret_cast<uint4*>(&f.u[4]) = *reinterpret_cast<const uint4*>(r + 8);
  return f.v;
}

// B fragment from an LDS-staged 64x32 chunk laid out row-major (e-row, k32).
__device__ __forceinline__ v16bf load_b_frag_lds(const __hip_bfloat16* bt,
                                                 int nt, int lane) {
  const int n  = lane & 15;
  const int kh = (lane >> 4) << 4;    // 0 or 16
  const __hip_bfloat16* r = bt + (nt * 16 + n) * 32 + kh;
  Frag16 f;
  *reinterpret_cast<uint4*>(&f.u[0]) = *reinterpret_cast<const uint4*>(r);
  *reinterpret_cast<uint4*>(&f.u[4]) = *reinterpret_cast<const uint4*>(r + 8);
  return f.v;
}

// ---------------------------------------------------------------------------
// prep: fp32 -> bf16 conversions (X, and W split into Wx | Wh per layer)
// ---------------------------------------------------------------------------
__global__ __launch_bounds__(256) void prep_kernel(
    const float* __restrict__ X, const float* __restrict__ W,
    __hip_bfloat16* __restrict__ Xbf,
    __hip_bfloat16* __restrict__ Wxbf, __hip_bfloat16* __restrict__ Whbf) {
  const size_t stride = (size_t)gridDim.x * blockDim.x;
  const size_t i0 = (size_t)blockIdx.x * blockDim.x + threadIdx.x;
  const size_t nX = (size_t)M_ * D_;
  for (size_t k = i0; k < nX; k += stride)
    Xbf[k] = __float2bfloat16(X[k]);
  const size_t nW = (size_t)NL_ * D_ * 2 * D_;
  for (size_t k = i0; k < nW; k += stride) {
    const size_t row = k / (2 * D_);          // row = layer*D + e
    const int    col = (int)(k - row * (2 * D_));
    const __hip_bfloat16 v = __float2bfloat16(W[k]);
    if (col < D_) Wxbf[row * D_ + col] = v;
    else          Whbf[row * D_ + (col - D_)] = v;
  }
}

// ---------------------------------------------------------------------------
// Phase 1: Xw[m,e] = sum_d Xbf[m,d] * Wx[e,d] + b[e]   (M=8192, N=K=1024)
// 8 waves/block, each one 16-row M tile; all waves share a 64-column e-slice.
// Wx chunks (64 x 32) are DMA'd into double-buffered LDS by the TDM (one wave
// issues; TENSORcnt tracks), overlapping the previous chunk's WMMAs.
// ---------------------------------------------------------------------------
__global__ __launch_bounds__(256, 1) void gemm_xw_kernel(
    const __hip_bfloat16* __restrict__ Xbf,
    const __hip_bfloat16* __restrict__ Wx,
    const float* __restrict__ bias,
    float* __restrict__ Xw) {
  __shared__ __align__(16) __hip_bfloat16 btile[2][64 * 32];  // 2 x 4 KB

  const int lane = threadIdx.x & 31;
  const int wave = threadIdx.x >> 5;
  const int m0 = (blockIdx.x * 8 + wave) * 16;
  const int e0 = blockIdx.y * 64;
  const int NCH = D_ / 32;   // 32 K-chunks

  if (wave == 0) {
    tdm_load_2d_bf16(lds_offset(&btile[0][0]), Wx + (size_t)e0 * D_,
                     /*tile*/32, 64, /*tensor*/D_, D_, /*stride*/D_);
  }

  v8f acc0 = {}, acc1 = {}, acc2 = {}, acc3 = {};
  v16bf a = load_a_frag(Xbf, D_, m0, 0, lane);

  for (int ic = 0; ic < NCH; ++ic) {
    if (wave == 0) __builtin_amdgcn_s_wait_tensorcnt(0);  // chunk ic landed
    __syncthreads();
    if (wave == 0 && ic + 1 < NCH) {
      tdm_load_2d_bf16(lds_offset(&btile[(ic + 1) & 1][0]),
                       Wx + (size_t)e0 * D_ + (ic + 1) * 32,
                       32, 64, D_, D_, D_);
    }
    // software-pipelined A fragment for the next chunk
    v16bf a_next = a;
    if (ic + 1 < NCH) {
      a_next = load_a_frag(Xbf, D_, m0, (ic + 1) * 32, lane);
      __builtin_prefetch(Xbf + (size_t)(m0 + (lane & 15)) * D_ + (ic + 1) * 32 + 32, 0, 3);
    }
    const __hip_bfloat16* bt = &btile[ic & 1][0];
    acc0 = wmma_bf16(a, load_b_frag_lds(bt, 0, lane), acc0);
    acc1 = wmma_bf16(a, load_b_frag_lds(bt, 1, lane), acc1);
    acc2 = wmma_bf16(a, load_b_frag_lds(bt, 2, lane), acc2);
    acc3 = wmma_bf16(a, load_b_frag_lds(bt, 3, lane), acc3);
    a = a_next;
    __syncthreads();   // reads of btile[ic&1] done before it is re-filled
  }

  // C/D layout: VGPR r holds M = r + 8*(lane>=16), N = lane&15
  const int n  = lane & 15;
  const int mh = (lane >> 4) * 8;
  v8f accs[4] = {acc0, acc1, acc2, acc3};
#pragma unroll
  for (int nt = 0; nt < 4; ++nt) {
    const int e = e0 + nt * 16 + n;
    const float bv = bias[e];
#pragma unroll
    for (int r = 0; r < 8; ++r)
      Xw[(size_t)(m0 + mh + r) * D_ + e] = accs[nt][r] + bv;
  }
}

// ---------------------------------------------------------------------------
// Grid-wide sense barrier (persistent scan kernel, 64 WGs all resident)
// ---------------------------------------------------------------------------
__device__ __forceinline__ void grid_barrier(int* count, int* sense, int nwg) {
  __syncthreads();
  if (threadIdx.x == 0) {
    const int my = atomicAdd(sense, 0);
    __threadfence();
    if (atomicAdd(count, 1) == nwg - 1) {
      atomicExch(count, 0);
      __threadfence();
      atomicAdd(sense, 1);
    } else {
      while (atomicAdd(sense, 0) == my) __builtin_amdgcn_s_sleep(1);
    }
  }
  __syncthreads();
}

// ---------------------------------------------------------------------------
// Phase 2: sequential scan. Each WG owns columns e0..e0+15; its 32 KB Wh slice
// is staged into LDS with one TDM DMA. K=1024 split across 8 waves (4 WMMAs
// each) + LDS reduce; grid barrier per timestep.
// ---------------------------------------------------------------------------
__global__ __launch_bounds__(256, 1) void rnn_scan_kernel(
    const __hip_bfloat16* __restrict__ Wh,   // (D, D) row-major (e, d), this layer
    float* __restrict__ XwH,                 // (B*L, D): in Xw, out fp32 H (aliases d_out)
    const float* __restrict__ h0,            // (B, D) initial hidden state
    __hip_bfloat16* __restrict__ hb0,
    __hip_bfloat16* __restrict__ hb1,
    __hip_bfloat16* __restrict__ Xbf,        // (B, L, D) bf16 activations for next layer
    int* count, int* sense) {
  __shared__ __align__(16) __hip_bfloat16 whs[16 * D_];  // 32 KB slice of Wh
  __shared__ float red[8][256];                          // 8 partial 16x16 tiles

  const int tid  = threadIdx.x;
  const int lane = tid & 31;
  const int wave = tid >> 5;
  const int e0   = blockIdx.x * 16;

  // One TDM DMA: 16 contiguous Wh rows (16*1024 bf16 = 32 KB) -> LDS.
  if (wave == 0) {
    tdm_load_2d_bf16(lds_offset(&whs[0]), Wh + (size_t)e0 * D_,
                     /*tile*/16 * D_, 1, /*tensor*/16 * D_, 1, /*stride*/16 * D_);
  }
  // Cooperative h init: 16*1024 elements over 64 WGs x 256 threads.
  {
    const int idx = blockIdx.x * 256 + tid;
    hb0[idx] = __float2bfloat16(h0[idx]);
  }
  if (wave == 0) __builtin_amdgcn_s_wait_tensorcnt(0);
  __syncthreads();
  grid_barrier(count, sense, SCAN_NWG);

  for (int t = 0; t < L_; ++t) {
    const __hip_bfloat16* hcur = (t & 1) ? hb1 : hb0;
    __hip_bfloat16* hnext      = (t & 1) ? hb0 : hb1;

    v8f acc = {};
    const int kbase = wave * 128;
#pragma unroll
    for (int i = 0; i < 4; ++i) {
      const int kk = kbase + i * 32;
      v16bf a = load_a_frag(hcur, D_, 0, kk, lane);
      v16bf b = load_b_frag(whs, D_, 0, kk, lane);    // LDS -> ds_load_b128
      acc = wmma_bf16(a, b, acc);
    }
    const int n  = lane & 15;
    const int mh = (lane >> 4) * 8;
#pragma unroll
    for (int r = 0; r < 8; ++r) red[wave][(mh + r) * 16 + n] = acc[r];
    __syncthreads();

    // Reduce the 8 K-partials, add Xw, tanh, write h / activations.
    {
      const int m  = tid >> 4;     // batch row 0..15
      const int nn = tid & 15;
      float s = 0.f;
#pragma unroll
      for (int w = 0; w < 8; ++w) s += red[w][tid];
      const size_t gi = ((size_t)m * L_ + t) * D_ + e0 + nn;
      const float hv = tanhf(s + XwH[gi]);
      XwH[gi] = hv;                               // fp32 (final output on last layer)
      hnext[m * D_ + e0 + nn] = __float2bfloat16(hv);
      Xbf[gi] = __float2bfloat16(hv);             // bf16 input for next layer's GEMM
    }
    grid_barrier(count, sense, SCAN_NWG);
  }
}

// ---------------------------------------------------------------------------
extern "C" void kernel_launch(void* const* d_in, const int* in_sizes, int n_in,
                              void* d_out, int out_size, void* d_ws, size_t ws_size,
                              hipStream_t stream) {
  const float* X   = (const float*)d_in[0];   // (B, L, D)
  const float* h0s = (const float*)d_in[1];   // (NL, B, D)
  const float* W   = (const float*)d_in[2];   // (NL, D, 2D)
  const float* bia = (const float*)d_in[3];   // (NL, D)
  float* out = (float*)d_out;                 // (B, L, D) fp32; doubles as Xw buffer

  char* ws = (char*)d_ws;
  size_t off = 0;
  __hip_bfloat16* Xbf  = (__hip_bfloat16*)(ws + off); off += (size_t)M_ * D_ * 2;         // 16 MB
  __hip_bfloat16* Wxbf = (__hip_bfloat16*)(ws + off); off += (size_t)NL_ * D_ * D_ * 2;   // 8 MB
  __hip_bfloat16* Whbf = (__hip_bfloat16*)(ws + off); off += (size_t)NL_ * D_ * D_ * 2;   // 8 MB
  __hip_bfloat16* hb0  = (__hip_bfloat16*)(ws + off); off += (size_t)B_ * D_ * 2;
  __hip_bfloat16* hb1  = (__hip_bfloat16*)(ws + off); off += (size_t)B_ * D_ * 2;
  int* counters        = (int*)(ws + off);            off += 256;

  // Deterministic barrier state each call (graph-capturable).
  (void)hipMemsetAsync(counters, 0, 256, stream);

  prep_kernel<<<2048, 256, 0, stream>>>(X, W, Xbf, Wxbf, Whbf);

  for (int layer = 0; layer < NL_; ++layer) {
    gemm_xw_kernel<<<dim3(M_ / 128, D_ / 64), 256, 0, stream>>>(
        Xbf, Wxbf + (size_t)layer * D_ * D_, bia + (size_t)layer * D_, out);
    rnn_scan_kernel<<<SCAN_NWG, 256, 0, stream>>>(
        Whbf + (size_t)layer * D_ * D_, out, h0s + (size_t)layer * B_ * D_,
        hb0, hb1, Xbf, counters, counters + 32);
  }
}
